// NN_88905823027230
// MI455X (gfx1250) — matrix operand order repaired and checked by reading, not compiled, not run on hardware
//
#include <hip/hip_runtime.h>
#include <math.h>

#define BDIM 256

typedef __attribute__((ext_vector_type(16))) _Float16 v16h;
typedef __attribute__((ext_vector_type(8)))  float    v8f;

#define B_     64
#define S_     21
#define BS_    1344   // B*S
#define SEC_   6
#define HID_   500
#define NN_    2
#define ATT_   128
#define OUT_   128
#define NPROTO_ 25
#define NCLS_  5

// ---------- helpers ----------
__device__ __forceinline__ float pe_val(int pos, int h, float d) {
  // pe[:,0::2]=sin(pos*div), pe[:,1::2]=cos(pos*div), div=exp(2i * -ln(1e4)/d)
  float i2  = (float)(h & ~1);
  float div = expf(i2 * (-9.210340371976184f / d));
  float ang = (float)pos * div;
  return (h & 1) ? cosf(ang) : sinf(ang);
}

__device__ __forceinline__ float wred(float v) {
#pragma unroll
  for (int off = 16; off > 0; off >>= 1) v += __shfl_xor(v, off, 32);
  return v;
}

// ---------- WMMA GEMM: C[M,N] = A[M,K] @ BT[N,K]^T (+bias, optional relu) ----------
// one wave32 per 32x32 C tile (2x2 fragments -> 4 wmma per K-step);
// A/B fragments use contiguous k order (k = 16*(lane/16) + element). Requires M%32==0 && N%32==0.
__global__ __launch_bounds__(32)
void k_gemm_wmma(const float* __restrict__ A, const float* __restrict__ BT,
                 const float* __restrict__ bias, float* __restrict__ C,
                 int M, int N, int K, int relu) {
  const int lane = threadIdx.x;
  const int n0 = blockIdx.x * 32;
  const int m0 = blockIdx.y * 32;
  const int l16 = lane & 15;
  const int grp = lane >> 4;
  const float* Arow0 = A  + (size_t)(m0 + l16) * K;
  const float* Arow1 = A  + (size_t)(m0 + 16 + l16) * K;
  const float* Brow0 = BT + (size_t)(n0 + l16) * K;
  const float* Brow1 = BT + (size_t)(n0 + 16 + l16) * K;
  v8f acc00 = {}, acc01 = {}, acc10 = {}, acc11 = {};
  int k0 = 0;
  for (; k0 + 32 <= K; k0 += 32) {      // full blocks: contiguous 128b loads
    v16h a0, a1, b0, b1;
    const float* ap0 = Arow0 + k0 + grp * 16;
    const float* ap1 = Arow1 + k0 + grp * 16;
#pragma unroll
    for (int i = 0; i < 8; ++i) {
      a0[i] = (_Float16)ap0[i]; a0[i + 8] = (_Float16)ap0[8 + i];
      a1[i] = (_Float16)ap1[i]; a1[i + 8] = (_Float16)ap1[8 + i];
    }
    const float* bp0 = Brow0 + k0 + grp * 16;
    const float* bp1 = Brow1 + k0 + grp * 16;
#pragma unroll
    for (int i = 0; i < 16; ++i) { b0[i] = (_Float16)bp0[i]; b1[i] = (_Float16)bp1[i]; }
    acc00 = __builtin_amdgcn_wmma_f32_16x16x32_f16(false, a0, false, b0, (short)0, acc00, false, false);
    acc01 = __builtin_amdgcn_wmma_f32_16x16x32_f16(false, a0, false, b1, (short)0, acc01, false, false);
    acc10 = __builtin_amdgcn_wmma_f32_16x16x32_f16(false, a1, false, b0, (short)0, acc10, false, false);
    acc11 = __builtin_amdgcn_wmma_f32_16x16x32_f16(false, a1, false, b1, (short)0, acc11, false, false);
  }
  if (k0 < K) {                          // K tail, predicated
    v16h a0, a1, b0, b1;
#pragma unroll
    for (int i = 0; i < 8; ++i) {
      int ka = k0 + grp * 16 + i;
      bool p0 = ka < K, p1 = (ka + 8) < K;
      a0[i]     = (_Float16)(p0 ? Arow0[ka]      : 0.0f);
      a0[i + 8] = (_Float16)(p1 ? Arow0[ka + 8] : 0.0f);
      a1[i]     = (_Float16)(p0 ? Arow1[ka]      : 0.0f);
      a1[i + 8] = (_Float16)(p1 ? Arow1[ka + 8] : 0.0f);
    }
#pragma unroll
    for (int i = 0; i < 16; ++i) {
      int kb = k0 + grp * 16 + i;
      bool p = kb < K;
      b0[i] = (_Float16)(p ? Brow0[kb] : 0.0f);
      b1[i] = (_Float16)(p ? Brow1[kb] : 0.0f);
    }
    acc00 = __builtin_amdgcn_wmma_f32_16x16x32_f16(false, a0, false, b0, (short)0, acc00, false, false);
    acc01 = __builtin_amdgcn_wmma_f32_16x16x32_f16(false, a0, false, b1, (short)0, acc01, false, false);
    acc10 = __builtin_amdgcn_wmma_f32_16x16x32_f16(false, a1, false, b0, (short)0, acc10, false, false);
    acc11 = __builtin_amdgcn_wmma_f32_16x16x32_f16(false, a1, false, b1, (short)0, acc11, false, false);
  }
  const int nA = n0 + l16;
  const int nB = n0 + 16 + l16;
  float bvA = bias ? bias[nA] : 0.0f;
  float bvB = bias ? bias[nB] : 0.0f;
#pragma unroll
  for (int r = 0; r < 8; ++r) {
    int mA = m0 + grp * 8 + r;
    int mB = m0 + 16 + grp * 8 + r;
    float v00 = acc00[r] + bvA;
    float v01 = acc01[r] + bvB;
    float v10 = acc10[r] + bvA;
    float v11 = acc11[r] + bvB;
    if (relu) {
      v00 = fmaxf(v00, 0.0f); v01 = fmaxf(v01, 0.0f);
      v10 = fmaxf(v10, 0.0f); v11 = fmaxf(v11, 0.0f);
    }
    C[(size_t)mA * N + nA] = v00;
    C[(size_t)mA * N + nB] = v01;
    C[(size_t)mB * N + nA] = v10;
    C[(size_t)mB * N + nB] = v11;
  }
}

// ---------- misc kernels ----------
__global__ void k_transpose(const float* __restrict__ src, float* __restrict__ dst, int K, int N) {
  int idx = blockIdx.x * blockDim.x + threadIdx.x;
  if (idx >= K * N) return;
  int k = idx / N, n = idx % N;
  dst[(size_t)n * K + k] = src[idx];
}

__global__ void k_xp(const float* __restrict__ x, float* __restrict__ xp) {
  int idx = blockIdx.x * blockDim.x + threadIdx.x;
  if (idx >= BS_ * SEC_ * HID_) return;
  int h = idx % HID_;
  int s = (idx / HID_) % SEC_;
  xp[idx] = x[idx] + pe_val(s, h, 500.0f);
}

__global__ void k_xsum(const float* __restrict__ xp, float* __restrict__ xsum) {
  int idx = blockIdx.x * blockDim.x + threadIdx.x;
  if (idx >= BS_ * HID_) return;
  int bsi = idx / HID_, h = idx % HID_;
  float s = 0.0f;
#pragma unroll
  for (int t = 0; t < SEC_; ++t) s += xp[(size_t)(bsi * SEC_ + t) * HID_ + h];
  xsum[idx] = s;
}

// logits[s] = (q[s] . ksum) / (sqrt(500)*6) + gumbel ; hard argmax + ST weight
__global__ void k_select(const float* __restrict__ q, const float* __restrict__ ksum,
                         const float* __restrict__ gum, int* __restrict__ sel,
                         float* __restrict__ won) {
  int idx = blockIdx.x * blockDim.x + threadIdx.x;
  if (idx >= BS_ * NN_) return;
  int bsi = idx / NN_;
  int n   = idx % NN_;
  const float scale = 1.0f / (sqrtf((float)HID_) * (float)SEC_);
  const float* kr = ksum + (size_t)bsi * (ATT_ * NN_) + n * ATT_;
  float y[SEC_];
  float ymax = -1e30f; int best = 0;
  for (int s = 0; s < SEC_; ++s) {
    const float* qr = q + (size_t)(bsi * SEC_ + s) * (ATT_ * NN_) + n * ATT_;
    float d = 0.0f;
    for (int h = 0; h < ATT_; ++h) d += qr[h] * kr[h];
    float v = d * scale + gum[idx * SEC_ + s];
    y[s] = v;
    if (v > ymax) { ymax = v; best = s; }
  }
  float den = 0.0f;
  for (int s = 0; s < SEC_; ++s) den += expf(y[s] - ymax);
  float ys = 1.0f / den;                 // softmax prob at argmax
  sel[idx] = best;
  won[idx] = (1.0f - ys) + ys;           // straight-through forward weight
}

__global__ void k_sample(const float* __restrict__ x, const int* __restrict__ sel,
                         const float* __restrict__ won, float* __restrict__ samp) {
  int idx = blockIdx.x * blockDim.x + threadIdx.x;
  if (idx >= BS_ * NN_ * HID_) return;
  int row = idx / HID_, h = idx % HID_;
  int bsi = row / NN_;
  samp[idx] = won[row] * x[(size_t)bsi * (SEC_ * HID_) + sel[row] * HID_ + h];
}

// fused conv1/pool4/bn -> conv2/pool5/bn -> conv3/pool4/bn (+flatten), all in LDS
__global__ void k_encoder(const float* __restrict__ samp,
                          const float* __restrict__ w1, const float* __restrict__ b1,
                          const float* __restrict__ g1, const float* __restrict__ bb1,
                          const float* __restrict__ w2, const float* __restrict__ b2,
                          const float* __restrict__ g2, const float* __restrict__ bb2,
                          const float* __restrict__ w3, const float* __restrict__ b3,
                          const float* __restrict__ g3, const float* __restrict__ bb3,
                          float* __restrict__ c3out) {
  __shared__ float sIn[HID_];
  __shared__ float sC1[32 * 125];
  __shared__ float sC2[64 * 25];
  const int bb  = blockIdx.x;
  const int tid = threadIdx.x;
  const float binv = rsqrtf(1.0f + 1e-5f);
  for (int i = tid; i < HID_; i += BDIM) sIn[i] = samp[(size_t)bb * HID_ + i];
  __syncthreads();
  // conv1 (1->32, k3 pad1) + relu + pool4 + bn
  for (int i = tid; i < 32 * 125; i += BDIM) {
    int ic = i / 125, ol = i % 125;
    float wv0 = w1[ic * 3], wv1 = w1[ic * 3 + 1], wv2 = w1[ic * 3 + 2];
    float bv = b1[ic], best = 0.0f;
#pragma unroll
    for (int p = 0; p < 4; ++p) {
      int l = 4 * ol + p;
      float s = bv + wv1 * sIn[l];
      if (l >= 1)        s += wv0 * sIn[l - 1];
      if (l <= HID_ - 2) s += wv2 * sIn[l + 1];
      best = fmaxf(best, fmaxf(s, 0.0f));
    }
    sC1[i] = best * (g1[ic] * binv) + bb1[ic];
  }
  __syncthreads();
  // conv2 (32->64, k3 pad1) + relu + pool5 + bn
  for (int i = tid; i < 64 * 25; i += BDIM) {
    int oc = i / 25, ol = i % 25;
    float bv = b2[oc], best = 0.0f;
    const float* wp = w2 + oc * 96;
    for (int p = 0; p < 5; ++p) {
      int l = 5 * ol + p;
      float s = bv;
      for (int ic = 0; ic < 32; ++ic) {
        const float* row = sC1 + ic * 125;
        float x0 = (l >= 1)   ? row[l - 1] : 0.0f;
        float x2 = (l <= 123) ? row[l + 1] : 0.0f;
        s += wp[ic * 3] * x0 + wp[ic * 3 + 1] * row[l] + wp[ic * 3 + 2] * x2;
      }
      best = fmaxf(best, fmaxf(s, 0.0f));
    }
    sC2[i] = best * (g2[oc] * binv) + bb2[oc];
  }
  __syncthreads();
  // conv3 (64->128, k3 pad1) + relu + pool4(floor 25/4=6) + bn, flattened (oc*6+ol)
  for (int i = tid; i < 128 * 6; i += BDIM) {
    int oc = i / 6, ol = i % 6;
    float bv = b3[oc], best = 0.0f;
    const float* wp = w3 + oc * 192;
    for (int p = 0; p < 4; ++p) {
      int l = 4 * ol + p;                // 0..23 (pos 24 dropped by floor-pool)
      float s = bv;
      for (int ic = 0; ic < 64; ++ic) {
        const float* row = sC2 + ic * 25;
        float x0 = (l >= 1) ? row[l - 1] : 0.0f;
        s += wp[ic * 3] * x0 + wp[ic * 3 + 1] * row[l] + wp[ic * 3 + 2] * row[l + 1];
      }
      best = fmaxf(best, fmaxf(s, 0.0f));
    }
    c3out[(size_t)bb * 768 + i] = best * (g3[oc] * binv) + bb3[oc];
  }
}

// prototype VQ: argmin_p ||z-p||^2, out = z + (p* - z); one wave per row
__global__ void k_proto(const float* __restrict__ z, const float* __restrict__ protos,
                        float* __restrict__ pst) {
  int gid = blockIdx.x * blockDim.x + threadIdx.x;
  int wid = gid >> 5, lane = gid & 31;
  if (wid >= BS_ * NN_) return;
  const float* zr = z + (size_t)wid * OUT_;
  float zv[4];
#pragma unroll
  for (int j = 0; j < 4; ++j) zv[j] = zr[lane * 4 + j];
  float zz = wred(zv[0]*zv[0] + zv[1]*zv[1] + zv[2]*zv[2] + zv[3]*zv[3]);
  float bestd = 1e30f; int bestp = 0;
  for (int p = 0; p < NPROTO_; ++p) {
    const float* pr = protos + p * OUT_ + lane * 4;
    float part = 0.0f;
#pragma unroll
    for (int j = 0; j < 4; ++j) { float pv = pr[j]; part += pv * pv - 2.0f * zv[j] * pv; }
    float d2 = zz + wred(part);
    if (d2 < bestd) { bestd = d2; bestp = p; }
  }
  const float* pr = protos + bestp * OUT_ + lane * 4;
  float* op = pst + (size_t)wid * OUT_ + lane * 4;
#pragma unroll
  for (int j = 0; j < 4; ++j) op[j] = zv[j] + (pr[j] - zv[j]);
}

__global__ void k_pclf(const float* __restrict__ pst, float* __restrict__ h) {
  int idx = blockIdx.x * blockDim.x + threadIdx.x;
  if (idx >= BS_ * OUT_) return;
  int row = idx / OUT_, c = idx % OUT_;
  int s = row % S_;
  h[idx] = (pst[(size_t)(row * 2) * OUT_ + c] + pst[(size_t)(row * 2 + 1) * OUT_ + c]) * 0.5f
           + pe_val(s, c, 128.0f);
}

// 4-head self-attention over S=21; one wave per (b,head), thread = query row
__global__ void k_attn(const float* __restrict__ qkv, float* __restrict__ ctx) {
  int b  = blockIdx.x >> 2;
  int hd = blockIdx.x & 3;
  int qi = threadIdx.x;
  if (qi >= S_) return;
  const float isc = rsqrtf(32.0f);
  float qv[32];
  const float* qr = qkv + (size_t)(b * S_ + qi) * 384 + hd * 32;
#pragma unroll
  for (int d = 0; d < 32; ++d) qv[d] = qr[d];
  float sc[S_]; float m = -1e30f;
  for (int k = 0; k < S_; ++k) {
    const float* kr = qkv + (size_t)(b * S_ + k) * 384 + 128 + hd * 32;
    float d = 0.0f;
#pragma unroll
    for (int e = 0; e < 32; ++e) d += qv[e] * kr[e];
    sc[k] = d * isc;
    m = fmaxf(m, sc[k]);
  }
  float den = 0.0f;
  for (int k = 0; k < S_; ++k) { sc[k] = expf(sc[k] - m); den += sc[k]; }
  float inv = 1.0f / den;
  float cv[32] = {};
  for (int k = 0; k < S_; ++k) {
    const float* vr = qkv + (size_t)(b * S_ + k) * 384 + 256 + hd * 32;
    float pk = sc[k] * inv;
#pragma unroll
    for (int e = 0; e < 32; ++e) cv[e] += pk * vr[e];
  }
  float* cr = ctx + (size_t)(b * S_ + qi) * OUT_ + hd * 32;
#pragma unroll
  for (int e = 0; e < 32; ++e) cr[e] = cv[e];
}

// out = LayerNorm(a + r)*g + beta ; one wave per 128-dim row
__global__ void k_addln(const float* __restrict__ a, const float* __restrict__ r,
                        const float* __restrict__ g, const float* __restrict__ beta,
                        float* __restrict__ out, int rows) {
  int gid = blockIdx.x * blockDim.x + threadIdx.x;
  int wid = gid >> 5, lane = gid & 31;
  if (wid >= rows) return;
  int base = wid * OUT_ + lane * 4;
  float v[4];
#pragma unroll
  for (int j = 0; j < 4; ++j) v[j] = a[base + j] + r[base + j];
  float mean = wred(v[0] + v[1] + v[2] + v[3]) * (1.0f / 128.0f);
  float p2 = 0.0f;
#pragma unroll
  for (int j = 0; j < 4; ++j) { float d = v[j] - mean; p2 += d * d; }
  float var = wred(p2) * (1.0f / 128.0f);
  float rs = rsqrtf(var + 1e-5f);
#pragma unroll
  for (int j = 0; j < 4; ++j) {
    int c = lane * 4 + j;
    out[base + j] = (v[j] - mean) * rs * g[c] + beta[c];
  }
}

__global__ void k_clf(const float* __restrict__ h2, const float* __restrict__ w,
                      const float* __restrict__ bias, float* __restrict__ out) {
  int idx = blockIdx.x * blockDim.x + threadIdx.x;
  if (idx >= BS_ * NCLS_) return;
  int row = idx / NCLS_, c = idx % NCLS_;
  float acc = bias[c];
  for (int d = 0; d < OUT_; ++d) acc += h2[(size_t)row * OUT_ + d] * w[d * NCLS_ + c];
  out[idx] = acc;
}

// ---------- launch ----------
extern "C" void kernel_launch(void* const* d_in, const int* in_sizes, int n_in,
                              void* d_out, int out_size, void* d_ws, size_t ws_size,
                              hipStream_t stream) {
  (void)in_sizes; (void)n_in; (void)out_size; (void)ws_size;
  const float* x      = (const float*)d_in[0];
  const float* gum    = (const float*)d_in[1];
  const float* wq     = (const float*)d_in[2];
  const float* wk     = (const float*)d_in[3];
  const float* w1     = (const float*)d_in[4];
  const float* b1     = (const float*)d_in[5];
  const float* g1     = (const float*)d_in[6];
  const float* bb1    = (const float*)d_in[7];
  const float* w2     = (const float*)d_in[8];
  const float* b2     = (const float*)d_in[9];
  const float* g2     = (const float*)d_in[10];
  const float* bb2    = (const float*)d_in[11];
  const float* w3     = (const float*)d_in[12];
  const float* b3     = (const float*)d_in[13];
  const float* g3     = (const float*)d_in[14];
  const float* bb3    = (const float*)d_in[15];
  const float* fc_w   = (const float*)d_in[16];
  const float* fc_b   = (const float*)d_in[17];
  const float* fc2_w  = (const float*)d_in[18];
  const float* fc2_b  = (const float*)d_in[19];
  const float* protos = (const float*)d_in[20];
  const float* in_w   = (const float*)d_in[21];
  const float* in_b   = (const float*)d_in[22];
  const float* out_w  = (const float*)d_in[23];
  const float* out_b  = (const float*)d_in[24];
  const float* ln1_g  = (const float*)d_in[25];
  const float* ln1_b  = (const float*)d_in[26];
  const float* ln2_g  = (const float*)d_in[27];
  const float* ln2_b  = (const float*)d_in[28];
  const float* ff1_w  = (const float*)d_in[29];
  const float* ff1_b  = (const float*)d_in[30];
  const float* ff2_w  = (const float*)d_in[31];
  const float* ff2_b  = (const float*)d_in[32];
  const float* clf_w  = (const float*)d_in[33];
  const float* clf_b  = (const float*)d_in[34];
  float* out = (float*)d_out;

  float* ws = (float*)d_ws;
  size_t o = 0;
  auto nxt = [&](size_t n) { float* p = ws + o; o += n; return p; };
  float* xp    = nxt(8064ull * 500);
  float* xsum  = nxt(1344ull * 500);
  float* qb    = nxt(8064ull * 256);
  float* ksum  = nxt(1344ull * 256);
  float* samp  = nxt(2688ull * 500);
  float* c3    = nxt(2688ull * 768);
  float* fco   = nxt(2688ull * 256);
  float* zb    = nxt(2688ull * 128);
  float* pst   = nxt(2688ull * 128);
  float* hb    = nxt(1344ull * 128);
  float* qkvb  = nxt(1344ull * 384);
  float* ctxb  = nxt(1344ull * 128);
  float* aproj = nxt(1344ull * 128);
  float* h1    = nxt(1344ull * 128);
  float* ffa   = nxt(1344ull * 128);
  float* ffbuf = nxt(1344ull * 128);
  float* h2    = nxt(1344ull * 128);
  float* wqT   = nxt(256ull * 500);
  float* wkT   = nxt(256ull * 500);
  float* fcT   = nxt(256ull * 768);
  float* fc2T  = nxt(128ull * 256);
  float* inT   = nxt(384ull * 128);
  float* outT  = nxt(128ull * 128);
  float* ff1T  = nxt(128ull * 128);
  float* ff2T  = nxt(128ull * 128);
  int*   selb  = (int*)nxt(2688);
  float* wonb  = nxt(2688);

  auto g1d = [](long n) { return dim3((unsigned)((n + BDIM - 1) / BDIM)); };
  dim3 blk(BDIM), wv(32);

  // weight transposes (BT layout for WMMA B-fragment contiguity)
  k_transpose<<<g1d(500 * 256), blk, 0, stream>>>(wq,    wqT,  500, 256);
  k_transpose<<<g1d(500 * 256), blk, 0, stream>>>(wk,    wkT,  500, 256);
  k_transpose<<<g1d(768 * 256), blk, 0, stream>>>(fc_w,  fcT,  768, 256);
  k_transpose<<<g1d(256 * 128), blk, 0, stream>>>(fc2_w, fc2T, 256, 128);
  k_transpose<<<g1d(128 * 384), blk, 0, stream>>>(in_w,  inT,  128, 384);
  k_transpose<<<g1d(128 * 128), blk, 0, stream>>>(out_w, outT, 128, 128);
  k_transpose<<<g1d(128 * 128), blk, 0, stream>>>(ff1_w, ff1T, 128, 128);
  k_transpose<<<g1d(128 * 128), blk, 0, stream>>>(ff2_w, ff2T, 128, 128);

  // hard-attention section selection
  k_xp  <<<g1d(BS_ * SEC_ * HID_), blk, 0, stream>>>(x, xp);
  k_xsum<<<g1d(BS_ * HID_),        blk, 0, stream>>>(xp, xsum);
  k_gemm_wmma<<<dim3(256 / 32, 8064 / 32), wv, 0, stream>>>(xp,   wqT, nullptr, qb,   8064, 256, 500, 0);
  k_gemm_wmma<<<dim3(256 / 32, 1344 / 32), wv, 0, stream>>>(xsum, wkT, nullptr, ksum, 1344, 256, 500, 0);
  k_select<<<g1d(BS_ * NN_), blk, 0, stream>>>(qb, ksum, gum, selb, wonb);
  k_sample<<<g1d(BS_ * NN_ * HID_), blk, 0, stream>>>(x, selb, wonb, samp);

  // fused conv encoder + FC head
  k_encoder<<<2688, blk, 0, stream>>>(samp, w1, b1, g1, bb1, w2, b2, g2, bb2,
                                      w3, b3, g3, bb3, c3);
  k_gemm_wmma<<<dim3(256 / 32, 2688 / 32), wv, 0, stream>>>(c3,  fcT,  fc_b,  fco, 2688, 256, 768, 0);
  k_gemm_wmma<<<dim3(128 / 32, 2688 / 32), wv, 0, stream>>>(fco, fc2T, fc2_b, zb,  2688, 128, 256, 0);

  // prototype layer + sequence encoder
  k_proto<<<g1d(2688 * 32), blk, 0, stream>>>(zb, protos, pst);
  k_pclf <<<g1d(BS_ * OUT_), blk, 0, stream>>>(pst, hb);
  k_gemm_wmma<<<dim3(384 / 32, 1344 / 32), wv, 0, stream>>>(hb, inT, in_b, qkvb, 1344, 384, 128, 0);
  k_attn<<<B_ * 4, wv, 0, stream>>>(qkvb, ctxb);
  k_gemm_wmma<<<dim3(128 / 32, 1344 / 32), wv, 0, stream>>>(ctxb, outT, out_b, aproj, 1344, 128, 128, 0);
  k_addln<<<g1d(1344 * 32), blk, 0, stream>>>(hb, aproj, ln1_g, ln1_b, h1, 1344);
  k_gemm_wmma<<<dim3(128 / 32, 1344 / 32), wv, 0, stream>>>(h1,  ff1T, ff1_b, ffa,   1344, 128, 128, 1);
  k_gemm_wmma<<<dim3(128 / 32, 1344 / 32), wv, 0, stream>>>(ffa, ff2T, ff2_b, ffbuf, 1344, 128, 128, 0);
  k_addln<<<g1d(1344 * 32), blk, 0, stream>>>(h1, ffbuf, ln2_g, ln2_b, h2, 1344);

  // classifier (N=5, not WMMA-shaped — trivial)
  k_clf<<<g1d(BS_ * NCLS_), blk, 0, stream>>>(h2, clf_w, clf_b, out);
}
